// STPWrapper_377957122320
// MI455X (gfx1250) — compile-verified, matrix operand order checked by reading, not run on hardware
//
#include <hip/hip_runtime.h>
#include <hip/hip_bf16.h>

// ---------------- problem constants (from reference) ----------------
#define NN     2048
#define TT     512
#define BB     32

#define DTC    1e-4f
#define U_C    0.3f
#define TAU_C  0.008f
#define TAUF_C 1.5f
#define TAUD_C 0.3f
#define ALPHA_C 1.5f
#define IB_C   8.0f
#define JEI_C  1.1f
#define JIE_C  2.2f

typedef __attribute__((ext_vector_type(8)))  __bf16 v8bf;
typedef __attribute__((ext_vector_type(16))) __bf16 v16bf;
typedef __attribute__((ext_vector_type(8)))  float  v8f;

// R(h) = ALPHA * softplus(h / ALPHA), numerically stable
__device__ __forceinline__ float rate_fn(float h) {
    float a  = h * (1.0f / ALPHA_C);
    float sp = fmaxf(a, 0.0f) + log1pf(__expf(-fabsf(a)));
    return ALPHA_C * sp;
}

// ---------------- K0a: convert J (constant across steps) -> bf16 ----------------
__global__ void convertJ_kernel(const float* __restrict__ J,
                                __bf16* __restrict__ Jbf, int total) {
    int i = blockIdx.x * blockDim.x + threadIdx.x;
    int stride = gridDim.x * blockDim.x;
    for (; i < total; i += stride) Jbf[i] = (__bf16)J[i];
}

// ---------------- K0b: initial A-matrix uxR = u0*x0*R(h0) = 0.3*1*R(0) ----------------
__global__ void init_uxr_kernel(__bf16* __restrict__ uxR) {
    const float R0 = rate_fn(0.0f);
    const __bf16 v = (__bf16)(U_C * 1.0f * R0);
    int i = (blockIdx.x * blockDim.x + threadIdx.x) * 4;  // 64 blocks * 256 thr * 4 = 65536
    #pragma unroll
    for (int j = 0; j < 4; ++j) uxR[i + j] = v;
}

// ---------------- fused per-step kernel ----------------
// 32 blocks x 256 threads = 256 wave32s: 2 M-tiles (batch) x 128 N-tiles of 16x16.
// Phase 1 (prologue): stage rate(h_I), inp into LDS; block 0 advances h_I from
//                     previous step's deterministic partial sums and emits outputs[t-1].
// Phase 2 (GEMM):     syn = uxR_in @ J^T via v_wmma_f32_16x16x32_bf16, K-loop over 2048.
// Phase 3 (update):   Euler update h/u/x, write outputs, emit NEXT step's A-matrix
//                     (uxR_out, double-buffered) + per-block partial sums of R(h2)
//                     and R(h2)*W_out (shfl_xor tree + LDS, fixed order, deterministic).
__global__ void step_fused_kernel(int t,
                                  const float* __restrict__ inp,    // [T,B]
                                  const float* __restrict__ W_in,   // [N]
                                  const float* __restrict__ W_out,  // [N]
                                  const __bf16* __restrict__ Jbf,   // [N,N]
                                  const __bf16* __restrict__ uxIn,  // [B,N] A for this step
                                  __bf16* __restrict__ uxOut,       // [B,N] A for next step
                                  const float* __restrict__ pRIn,   // [32 blk][32 b]
                                  const float* __restrict__ pOIn,
                                  float* __restrict__ pROut,
                                  float* __restrict__ pOOut,
                                  float* __restrict__ all_h,        // [T,B,N]
                                  float* __restrict__ all_u,
                                  float* __restrict__ all_x,
                                  float* __restrict__ all_hI,       // [T,B]
                                  float* __restrict__ outputs)      // [T,B]
{
    __shared__ float sRI[BB];        // rate(h_I entering this step), per batch
    __shared__ float sInp[BB];       // inp[t, b]
    __shared__ float sRedR[8][16];   // per-wave per-local-batch partial sums
    __shared__ float sRedO[8][16];

    const int tid  = threadIdx.x;
    const int lane = tid & 31;
    const int wave = tid >> 5;
    const int hi   = lane >> 4;
    const int l15  = lane & 15;
    const int m_tile = wave >> 2;                          // 0..1
    const int n0     = (blockIdx.x * 4 + (wave & 3)) * 16; // 0..2032

    // ---- Phase 1: prologue ----
    if (tid < BB) {
        float hI = (t == 0) ? 0.0f : all_hI[(size_t)(t - 1) * BB + tid];
        sRI[tid]  = rate_fn(hI);
        sInp[tid] = inp[(size_t)t * BB + tid];
        // block 0 only: advance h_I from previous step's deterministic partial
        // sums and emit outputs[t-1] (fixed summation order -> deterministic).
        if (blockIdx.x == 0) {
            float sumR, sumO = 0.0f;
            if (t == 0) {
                sumR = (float)NN * rate_fn(0.0f);
            } else {
                sumR = 0.0f;
                for (int blk = 0; blk < BB; ++blk) {
                    sumR += pRIn[blk * BB + tid];
                    sumO += pOIn[blk * BB + tid];
                }
            }
            all_hI[(size_t)t * BB + tid] =
                hI + DTC * ((-hI + JIE_C * sumR) * (1.0f / TAU_C));
            if (t > 0) outputs[(size_t)(t - 1) * BB + tid] = sumO;
        }
    }
    __syncthreads();

    // ---- Phase 2: WMMA GEMM  syn[b,n] = sum_k uxIn[b,k] * J[n,k] ----
    const __bf16* aRow = uxIn + (size_t)(m_tile * 16 + l15) * NN;
    const __bf16* bRow = Jbf  + (size_t)(n0 + l15) * NN;

    v8f acc = {};
    for (int k0 = 0; k0 < NN; k0 += 32) {
        const int c0 = k0 + hi * 8;
        const int c1 = k0 + 16 + hi * 8;
        v8bf a0 = *(const v8bf*)(aRow + c0);
        v8bf a1 = *(const v8bf*)(aRow + c1);
        v8bf b0 = *(const v8bf*)(bRow + c0);
        v8bf b1 = *(const v8bf*)(bRow + c1);
        v16bf A  = __builtin_shufflevector(a0, a1, 0,1,2,3,4,5,6,7,8,9,10,11,12,13,14,15);
        v16bf Bm = __builtin_shufflevector(b0, b1, 0,1,2,3,4,5,6,7,8,9,10,11,12,13,14,15);
        acc = __builtin_amdgcn_wmma_f32_16x16x32_bf16(
                  false, A, false, Bm, (short)0, acc, false, false);
    }

    // ---- Phase 3: fused Euler update + next-step A-matrix + partial sums ----
    const size_t curBase  = (size_t)t * BB * NN;
    const size_t prevBase = (size_t)(t - 1) * BB * NN;
    const int n = n0 + l15;
    const float win  = W_in[n];
    const float wout = W_out[n];

    float pr[8], po[8];
    #pragma unroll
    for (int i = 0; i < 8; ++i) {
        const int b = m_tile * 16 + (i + 8 * hi);
        const size_t poff = prevBase + (size_t)b * NN + n;
        float h = (t == 0) ? 0.0f : all_h[poff];
        float u = (t == 0) ? U_C  : all_u[poff];
        float x = (t == 0) ? 1.0f : all_x[poff];

        float syn = acc[i];
        float R   = rate_fn(h);
        float Ie  = sInp[b] * win;
        float RI  = sRI[b];

        float dh = (-h + syn + IB_C + Ie) * (1.0f / TAU_C) - (JEI_C / TAU_C) * RI;
        float du = (U_C - u) * (1.0f / TAUF_C) + U_C * (1.0f - u) * R;
        float dx = (1.0f - x) * (1.0f / TAUD_C) - u * x * R;

        float h2 = h + DTC * dh;
        float u2 = u + DTC * du;
        float x2 = x + DTC * dx;

        const size_t o = curBase + (size_t)b * NN + n;
        all_h[o] = h2;
        all_u[o] = u2;
        all_x[o] = x2;

        float R2 = rate_fn(h2);
        uxOut[(size_t)b * NN + n] = (__bf16)(u2 * x2 * R2);  // next step's A
        pr[i] = R2;
        po[i] = R2 * wout;
    }

    // deterministic per-batch reduction over this block's 64 n-columns:
    // shfl_xor tree over the 16 lanes sharing a batch, then fixed-order LDS combine.
    #pragma unroll
    for (int i = 0; i < 8; ++i) {
        float r = pr[i], o = po[i];
        #pragma unroll
        for (int m = 1; m < 16; m <<= 1) {
            r += __shfl_xor(r, m, 32);
            o += __shfl_xor(o, m, 32);
        }
        if (l15 == 0) {                       // lanes 0 and 16
            sRedR[wave][i + 8 * hi] = r;
            sRedO[wave][i + 8 * hi] = o;
        }
    }
    __syncthreads();
    if (tid < BB) {
        const int b = tid;
        const int wb = (b >> 4) * 4;          // waves covering this batch's m-tile
        const int bl = b & 15;
        float r = sRedR[wb][bl] + sRedR[wb + 1][bl] + sRedR[wb + 2][bl] + sRedR[wb + 3][bl];
        float o = sRedO[wb][bl] + sRedO[wb + 1][bl] + sRedO[wb + 2][bl] + sRedO[wb + 3][bl];
        pROut[blockIdx.x * BB + b] = r;
        pOOut[blockIdx.x * BB + b] = o;
    }
}

// ---------------- epilogue: outputs[T-1] from final partial sums ----------------
__global__ void finish_kernel(const float* __restrict__ pO,
                              float* __restrict__ outputs) {
    int b = threadIdx.x;                      // 32 threads
    float o = 0.0f;
    for (int blk = 0; blk < BB; ++blk) o += pO[blk * BB + b];
    outputs[(size_t)(TT - 1) * BB + b] = o;
}

// ---------------- driver ----------------
extern "C" void kernel_launch(void* const* d_in, const int* in_sizes, int n_in,
                              void* d_out, int out_size, void* d_ws, size_t ws_size,
                              hipStream_t stream) {
    const float* inp   = (const float*)d_in[0]; // [T,B,1]
    const float* W_in  = (const float*)d_in[1]; // [N,1]
    const float* J     = (const float*)d_in[2]; // [N,N]
    const float* W_out = (const float*)d_in[3]; // [1,N]

    const size_t SLAB = (size_t)TT * BB * NN;
    float* all_h   = (float*)d_out;
    float* all_u   = all_h + SLAB;
    float* all_x   = all_u + SLAB;
    float* all_hI  = all_x + SLAB;
    float* outputs = all_hI + (size_t)TT * BB;

    // workspace: Jbf (8MB) | uxR ping-pong (2x128KB) | partial sums ping-pong (4x4KB)
    __bf16* Jbf  = (__bf16*)d_ws;
    __bf16* ux0  = Jbf + (size_t)NN * NN;
    __bf16* ux1  = ux0 + (size_t)BB * NN;
    float*  pR0  = (float*)(ux1 + (size_t)BB * NN);
    float*  pR1  = pR0 + BB * BB;
    float*  pO0  = pR1 + BB * BB;
    float*  pO1  = pO0 + BB * BB;
    __bf16* uxbuf[2] = {ux0, ux1};
    float*  pRbuf[2] = {pR0, pR1};
    float*  pObuf[2] = {pO0, pO1};
    (void)in_sizes; (void)n_in; (void)out_size; (void)ws_size;

    convertJ_kernel<<<2048, 256, 0, stream>>>(J, Jbf, NN * NN);
    init_uxr_kernel<<<64, 256, 0, stream>>>(ux0);   // buffer 0 = A-matrix for t=0

    for (int t = 0; t < TT; ++t) {
        const int rd = t & 1, wr = (t + 1) & 1;
        step_fused_kernel<<<BB, 256, 0, stream>>>(t, inp, W_in, W_out, Jbf,
                                                  uxbuf[rd], uxbuf[wr],
                                                  pRbuf[rd], pObuf[rd],
                                                  pRbuf[wr], pObuf[wr],
                                                  all_h, all_u, all_x,
                                                  all_hI, outputs);
    }
    // outputs[T-1] from partials written by step T-1 (buffer (T)&1 == 0)
    finish_kernel<<<1, BB, 0, stream>>>(pObuf[TT & 1], outputs);
}
